// TransformerAttention_652835029184
// MI455X (gfx1250) — matrix operand order repaired and checked by reading, not compiled, not run on hardware
//
#include <hip/hip_runtime.h>
#include <hip/hip_bf16.h>

typedef __attribute__((ext_vector_type(16))) __bf16 v16bf;
typedef __attribute__((ext_vector_type(8)))  float  v8f;
typedef unsigned short ushort_t;

#define HEADS 16
#define DHEAD 64
#define BATCH 4
#define SEQ   2048
#define DIM   1024
#define NTOK  (BATCH*SEQ)   /* 8192 */

__device__ __forceinline__ ushort_t f2bf(float f) {
  union { float f; unsigned u; } v; v.f = f;
  unsigned u = v.u;
  unsigned r = (u + 0x7FFFu + ((u >> 16) & 1u)) >> 16;  // round-to-nearest-even
  return (ushort_t)r;
}

__device__ __forceinline__ v8f wmma_bf16(v16bf a, v16bf b, v8f c) {
  return __builtin_amdgcn_wmma_f32_16x16x32_bf16(false, a, false, b, (short)0, c,
                                                 false, false);
}

// ---------- weight transpose + f32->bf16 convert: out[n*K+k] = in[k*N+n] ----------
__global__ void k_transpose_cvt(const float* __restrict__ in, ushort_t* __restrict__ out,
                                int K, int N) {
  int idx = blockIdx.x * 256 + threadIdx.x;
  if (idx >= K * N) return;
  int k = idx / N, n = idx % N;
  out[(size_t)n * K + k] = f2bf(in[idx]);
}

// ---------- layernorm -> bf16 xn; also raw x -> bf16 ----------
__global__ void k_layernorm(const float* __restrict__ x, const float* __restrict__ gamma,
                            ushort_t* __restrict__ xn, ushort_t* __restrict__ xb) {
  __shared__ float s1[256], s2[256];
  int row = blockIdx.x, t = threadIdx.x;
  const float4 v = ((const float4*)(x + (size_t)row * DIM))[t];
  s1[t] = v.x + v.y + v.z + v.w;
  s2[t] = v.x * v.x + v.y * v.y + v.z * v.z + v.w * v.w;
  __syncthreads();
  for (int o = 128; o > 0; o >>= 1) {
    if (t < o) { s1[t] += s1[t + o]; s2[t] += s2[t + o]; }
    __syncthreads();
  }
  float mu  = s1[0] * (1.f / DIM);
  float var = s2[0] * (1.f / DIM) - mu * mu;
  float inv = rsqrtf(var + 1e-5f);
  const float4 g = ((const float4*)gamma)[t];
  float y0 = (v.x - mu) * inv * g.x, y1 = (v.y - mu) * inv * g.y;
  float y2 = (v.z - mu) * inv * g.z, y3 = (v.w - mu) * inv * g.w;
  uint2 pn, pr;
  pn.x = f2bf(y0) | ((unsigned)f2bf(y1) << 16);
  pn.y = f2bf(y2) | ((unsigned)f2bf(y3) << 16);
  pr.x = f2bf(v.x) | ((unsigned)f2bf(v.y) << 16);
  pr.y = f2bf(v.z) | ((unsigned)f2bf(v.w) << 16);
  ((uint2*)(xn + (size_t)row * DIM))[t] = pn;
  ((uint2*)(xb + (size_t)row * DIM))[t] = pr;
}

// ---------- pipelined bf16 GEMM: 32x64 tile per wave, C = A[M,K] * Bt[N,K]^T ----------
struct KTile { v16bf a0, a1, b0, b1, b2, b3; };

__device__ __forceinline__ void load_ktile(KTile& t,
                                           const ushort_t* __restrict__ arow0,
                                           const ushort_t* __restrict__ arow1,
                                           const ushort_t* __restrict__ bbase,
                                           int k0, int K, int half, int l16) {
  // A 16x32 bf16 layout: lanes 0-15 K=[0..15], lanes 16-31 K=[16..31]
  const __bf16* p00 = (const __bf16*)(arow0 + k0 + half * 16);
  const __bf16* p01 = (const __bf16*)(arow0 + k0 + half * 16 + 8);
  const __bf16* p10 = (const __bf16*)(arow1 + k0 + half * 16);
  const __bf16* p11 = (const __bf16*)(arow1 + k0 + half * 16 + 8);
#pragma unroll
  for (int i = 0; i < 8; i++) {
    t.a0[i] = p00[i]; t.a0[8 + i] = p01[i];
    t.a1[i] = p10[i]; t.a1[8 + i] = p11[i];
  }
  // B 32x16: lane n=l16, K = half*16 + i (contiguous 32B per lane in Bt row)
  const ushort_t* bp = bbase + (size_t)l16 * K + k0 + half * 16;
  t.b0 = *(const v16bf*)(bp);
  t.b1 = *(const v16bf*)(bp + (size_t)16 * K);
  t.b2 = *(const v16bf*)(bp + (size_t)32 * K);
  t.b3 = *(const v16bf*)(bp + (size_t)48 * K);
}

template <bool OUTF32>
__global__ __launch_bounds__(32, 1)
void k_gemm(const ushort_t* __restrict__ A, const ushort_t* __restrict__ Bt,
            void* __restrict__ Cv, int K, int ldc, float scale) {
  const int lane = threadIdx.x;
  const int half = lane >> 4, l16 = lane & 15;
  const int mb = blockIdx.x * 32;
  const int nb = blockIdx.y * 64;
  v8f acc[2][4] = {};
  const ushort_t* arow0 = A + (size_t)(mb + l16) * K;
  const ushort_t* arow1 = A + (size_t)(mb + 16 + l16) * K;
  const ushort_t* bbase = Bt + (size_t)nb * K;

  KTile t0, t1;
  load_ktile(t0, arow0, arow1, bbase, 0, K, half, l16);
  for (int k0 = 0; k0 < K; k0 += 64) {   // K is a multiple of 64
    if (k0 + 32 < K) load_ktile(t1, arow0, arow1, bbase, k0 + 32, K, half, l16);
    acc[0][0] = wmma_bf16(t0.a0, t0.b0, acc[0][0]);
    acc[0][1] = wmma_bf16(t0.a0, t0.b1, acc[0][1]);
    acc[0][2] = wmma_bf16(t0.a0, t0.b2, acc[0][2]);
    acc[0][3] = wmma_bf16(t0.a0, t0.b3, acc[0][3]);
    acc[1][0] = wmma_bf16(t0.a1, t0.b0, acc[1][0]);
    acc[1][1] = wmma_bf16(t0.a1, t0.b1, acc[1][1]);
    acc[1][2] = wmma_bf16(t0.a1, t0.b2, acc[1][2]);
    acc[1][3] = wmma_bf16(t0.a1, t0.b3, acc[1][3]);
    if (k0 + 64 < K) load_ktile(t0, arow0, arow1, bbase, k0 + 64, K, half, l16);
    acc[0][0] = wmma_bf16(t1.a0, t1.b0, acc[0][0]);
    acc[0][1] = wmma_bf16(t1.a0, t1.b1, acc[0][1]);
    acc[0][2] = wmma_bf16(t1.a0, t1.b2, acc[0][2]);
    acc[0][3] = wmma_bf16(t1.a0, t1.b3, acc[0][3]);
    acc[1][0] = wmma_bf16(t1.a1, t1.b0, acc[1][0]);
    acc[1][1] = wmma_bf16(t1.a1, t1.b1, acc[1][1]);
    acc[1][2] = wmma_bf16(t1.a1, t1.b2, acc[1][2]);
    acc[1][3] = wmma_bf16(t1.a1, t1.b3, acc[1][3]);
  }
#pragma unroll
  for (int g = 0; g < 2; g++) {
#pragma unroll
    for (int c = 0; c < 4; c++) {
#pragma unroll
      for (int r = 0; r < 8; r++) {
        int row = mb + g * 16 + half * 8 + r;   // C layout: M = r + 8*(lane>=16)
        int col = nb + c * 16 + l16;
        float v = acc[g][c][r] * scale;
        if (OUTF32) ((float*)Cv)[(size_t)row * ldc + col] = v;
        else        ((ushort_t*)Cv)[(size_t)row * ldc + col] = f2bf(v);
      }
    }
  }
}

// ---------- V transpose: vt[b][d][j] = kv[b*SEQ+j][64+d] ----------
__global__ void k_transpose_v(const ushort_t* __restrict__ kv, ushort_t* __restrict__ vt) {
  int idx = blockIdx.x * 256 + threadIdx.x;   // B*64*SEQ total
  int b = idx / (DHEAD * SEQ);
  int rem = idx % (DHEAD * SEQ);
  int d = rem / SEQ, j = rem % SEQ;
  vt[idx] = kv[((size_t)b * SEQ + j) * 128 + DHEAD + d];
}

// ---------- flash attention, 16 query rows per wave, 32 keys per step ----------
__global__ __launch_bounds__(32, 1)
void k_attention(const ushort_t* __restrict__ Q, const ushort_t* __restrict__ KV,
                 const ushort_t* __restrict__ VT, const float* __restrict__ bias,
                 const unsigned char* __restrict__ mask, ushort_t* __restrict__ AO) {
  __shared__ __bf16 pbuf[16 * 32];
  const int lane = threadIdx.x;
  const int half = lane >> 4, l16 = lane & 15;
  const int qbase = blockIdx.x * 16;
  const int h = blockIdx.y;
  const int b = blockIdx.z;

  // A operand for S: 16 query rows x d=64 (already pre-scaled by DIM_HEAD^-0.5)
  const __bf16* qrow = (const __bf16*)(Q + ((size_t)b * SEQ + qbase + l16) * DIM + h * DHEAD);
  v16bf qa0, qa1;
#pragma unroll
  for (int i = 0; i < 8; i++) {
    qa0[i]     = qrow[half * 16 + i];
    qa0[8 + i] = qrow[half * 16 + 8 + i];
    qa1[i]     = qrow[32 + half * 16 + i];
    qa1[8 + i] = qrow[32 + half * 16 + 8 + i];
  }

  v8f o[4] = {};
  float m_run[8], l_run[8];
#pragma unroll
  for (int r = 0; r < 8; r++) { m_run[r] = -3.0e38f; l_run[r] = 0.f; }

  const ushort_t* Kb = KV + (size_t)b * SEQ * 128;          // K at col 0 of kv
  const __bf16*  Vb = (const __bf16*)(VT + (size_t)b * DHEAD * SEQ);
  const float*   bh = bias + (size_t)h * SEQ * SEQ;
  const unsigned char* mk = mask + (size_t)b * SEQ;
  const int row0 = qbase + half * 8;

  for (int jb = 0; jb <= qbase + 15; jb += 32) {
    // ---- S = Q K^T for 32 keys (two 16x16 tiles, K=64 split into 2 WMMAs each)
    v8f s0 = {}, s1 = {};
    {
      const __bf16* kr0 = (const __bf16*)(Kb + (size_t)(jb + l16) * 128);
      s0 = wmma_bf16(qa0, *(const v16bf*)(kr0 + half * 16), s0);
      s0 = wmma_bf16(qa1, *(const v16bf*)(kr0 + 32 + half * 16), s0);
      const __bf16* kr1 = (const __bf16*)(Kb + (size_t)(jb + 16 + l16) * 128);
      s1 = wmma_bf16(qa0, *(const v16bf*)(kr1 + half * 16), s1);
      s1 = wmma_bf16(qa1, *(const v16bf*)(kr1 + 32 + half * 16), s1);
    }
    // ---- prefetch next step's K rows and bias cachelines (bias is the HBM roofline)
    if (jb + 32 <= qbase + 15) {
      __builtin_prefetch(Kb + (size_t)(jb + 32 + l16) * 128, 0, 3);
      __builtin_prefetch(Kb + (size_t)(jb + 48 + l16) * 128, 0, 3);
      const int pc = jb + 32 + l16;
      if (pc < SEQ) {
#pragma unroll
        for (int r = 0; r < 8; r++)
          __builtin_prefetch(&bh[(size_t)(row0 + r) * SEQ + pc], 0, 3);
      }
    }
    // ---- bias + key mask + causal mask (C layout: row = qbase + 8*half + r, col = l16)
    const int c0 = jb + l16;
    const int c1 = jb + 16 + l16;
    const bool km0 = mk[c0] != 0;
    const bool km1 = mk[c1] != 0;
#pragma unroll
    for (int r = 0; r < 8; r++) {
      const int row = row0 + r;
      float v0 = s0[r] + bh[(size_t)row * SEQ + c0];
      float v1 = s1[r] + bh[(size_t)row * SEQ + c1];
      s0[r] = (km0 && c0 <= row) ? v0 : -3.0e38f;
      s1[r] = (km1 && c1 <= row) ? v1 : -3.0e38f;
    }
    // ---- online softmax: per-row stats live in VGPR index r, reduced over 16 lanes
#pragma unroll
    for (int r = 0; r < 8; r++) {
      float mx = fmaxf(s0[r], s1[r]);
      mx = fmaxf(mx, __shfl_xor(mx, 1));
      mx = fmaxf(mx, __shfl_xor(mx, 2));
      mx = fmaxf(mx, __shfl_xor(mx, 4));
      mx = fmaxf(mx, __shfl_xor(mx, 8));
      float mnew = fmaxf(m_run[r], mx);
      float corr = __expf(m_run[r] - mnew);
      float e0 = __expf(s0[r] - mnew);
      float e1 = __expf(s1[r] - mnew);
      float rs = e0 + e1;
      rs += __shfl_xor(rs, 1);
      rs += __shfl_xor(rs, 2);
      rs += __shfl_xor(rs, 4);
      rs += __shfl_xor(rs, 8);
      l_run[r] = l_run[r] * corr + rs;
      m_run[r] = mnew;
      s0[r] = e0; s1[r] = e1;
      o[0][r] *= corr; o[1][r] *= corr; o[2][r] *= corr; o[3][r] *= corr;
    }
    // ---- repack P (C layout) -> A layout via LDS
#pragma unroll
    for (int r = 0; r < 8; r++) {
      pbuf[(half * 8 + r) * 32 + l16]      = (__bf16)s0[r];
      pbuf[(half * 8 + r) * 32 + 16 + l16] = (__bf16)s1[r];
    }
    __syncthreads();
    v16bf pa;
#pragma unroll
    for (int i = 0; i < 8; i++) {
      pa[i]     = pbuf[l16 * 32 + half * 16 + i];
      pa[8 + i] = pbuf[l16 * 32 + half * 16 + 8 + i];
    }
    __syncthreads();
    // ---- O += P * V (V^T rows are lane-contiguous in the key dimension)
#pragma unroll
    for (int c = 0; c < 4; c++) {
      v16bf vb = *(const v16bf*)(Vb + (size_t)(c * 16 + l16) * SEQ + jb + half * 16);
      o[c] = wmma_bf16(pa, vb, o[c]);
    }
  }
  // ---- epilogue: O / l, write bf16 into [b*n, h*64 .. h*64+63]
  ushort_t* orow = AO + ((size_t)b * SEQ + qbase) * DIM + h * DHEAD;
#pragma unroll
  for (int r = 0; r < 8; r++) {
    float inv = 1.f / l_run[r];
    ushort_t* op = orow + (size_t)(half * 8 + r) * DIM;
#pragma unroll
    for (int c = 0; c < 4; c++)
      op[c * 16 + l16] = f2bf(o[c][r] * inv);
  }
}

extern "C" void kernel_launch(void* const* d_in, const int* in_sizes, int n_in,
                              void* d_out, int out_size, void* d_ws, size_t ws_size,
                              hipStream_t stream) {
  const float*         x     = (const float*)d_in[0];
  const unsigned char* mask  = (const unsigned char*)d_in[1];
  const float*         bias  = (const float*)d_in[2];
  const float*         gamma = (const float*)d_in[3];
  const float*         w_q   = (const float*)d_in[4];
  const float*         w_kv  = (const float*)d_in[5];
  const float*         w_out = (const float*)d_in[6];
  (void)in_sizes; (void)n_in; (void)out_size; (void)ws_size;

  ushort_t* p    = (ushort_t*)d_ws;
  ushort_t* wqT  = p; p += (size_t)DIM * DIM;
  ushort_t* wkvT = p; p += (size_t)128 * DIM;
  ushort_t* woT  = p; p += (size_t)DIM * DIM;
  ushort_t* xn   = p; p += (size_t)NTOK * DIM;
  ushort_t* xb   = p; p += (size_t)NTOK * DIM;
  ushort_t* qb   = p; p += (size_t)NTOK * DIM;
  ushort_t* kvb  = p; p += (size_t)NTOK * 128;
  ushort_t* vt   = p; p += (size_t)BATCH * DHEAD * SEQ;
  ushort_t* ao   = p; p += (size_t)NTOK * DIM;

  k_transpose_cvt<<<(DIM * DIM + 255) / 256, 256, 0, stream>>>(w_q, wqT, DIM, DIM);
  k_transpose_cvt<<<(DIM * 128 + 255) / 256, 256, 0, stream>>>(w_kv, wkvT, DIM, 128);
  k_transpose_cvt<<<(DIM * DIM + 255) / 256, 256, 0, stream>>>(w_out, woT, DIM, DIM);
  k_layernorm<<<NTOK, 256, 0, stream>>>(x, gamma, xn, xb);

  const float scale = 0.125f;  // DIM_HEAD^-0.5
  k_gemm<false><<<dim3(NTOK / 32, DIM / 64), 32, 0, stream>>>(xn, wqT, qb, DIM, DIM, scale);
  k_gemm<false><<<dim3(NTOK / 32, 128 / 64), 32, 0, stream>>>(xb, wkvT, kvb, DIM, 128, 1.0f);
  k_transpose_v<<<(BATCH * DHEAD * SEQ + 255) / 256, 256, 0, stream>>>(kvb, vt);
  k_attention<<<dim3(SEQ / 16, HEADS, BATCH), 32, 0, stream>>>(qb, kvb, vt, bias, mask, ao);
  k_gemm<true><<<dim3(NTOK / 32, DIM / 64), 32, 0, stream>>>(ao, woT, d_out, DIM, DIM, 1.0f);
}
